// SelfAttention_v2_70282844832521
// MI455X (gfx1250) — matrix operand order repaired and checked by reading, not compiled, not run on hardware
//
#include <hip/hip_runtime.h>
#include <math.h>

// Problem sizes (fixed by the reference)
#define N_TOK 8192
#define D_DIM 1024
#define SCALE 0.03125f /* 1/sqrt(1024) */

typedef __attribute__((ext_vector_type(16))) __bf16 v16bf;
typedef __attribute__((ext_vector_type(8)))  float  v8f;
typedef __attribute__((ext_vector_type(8)))  unsigned short ushort8;

union FragU { ushort8 h[2]; v16bf v; };

__device__ __forceinline__ unsigned short f2bf(float f) {
  unsigned int u = __float_as_uint(f);
  u += 0x7FFFu + ((u >> 16) & 1u); // round-to-nearest-even
  return (unsigned short)(u >> 16);
}

__device__ __forceinline__ v8f wmma_bf16(v16bf a, v16bf b, v8f c) {
  // (neg_a, A, neg_b, B, c_mod, C, reuse_a, reuse_b)
  return __builtin_amdgcn_wmma_f32_16x16x32_bf16(false, a, false, b, (short)0, c,
                                                 false, false);
}

// A-fragment (16x32 bf16) from a row-major [*, stride] bf16 array.
// Lane layout: row M=lane%16, K = k0 + hi*16 .. +15 (contiguous, matching
// load_b_frag) -> two contiguous 16B loads.
__device__ __forceinline__ v16bf load_a_frag(const unsigned short* base, int stride,
                                             int row, int k0, int hi) {
  FragU u;
  u.h[0] = *(const ushort8*)(base + row * stride + k0 + hi * 16);
  u.h[1] = *(const ushort8*)(base + row * stride + k0 + hi * 16 + 8);
  return u.v;
}

// B-fragment (32x16 bf16). colbase points at element (K=k0, N=this lane's col)
// in a layout where K is contiguous per column. Lanes 0-15: K=k0..k0+15,
// lanes 16-31: K=k0+16..k0+31 -> two contiguous 16B loads.
__device__ __forceinline__ v16bf load_b_frag(const unsigned short* colbase, int hi) {
  FragU u;
  const ushort8* p = (const ushort8*)(colbase + hi * 16);
  u.h[0] = p[0];
  u.h[1] = p[1];
  return u.v;
}

// ---------------------------------------------------------------- converts
__global__ void cvt_f32_bf16(const float* __restrict__ in,
                             unsigned short* __restrict__ out, int n) {
  int i = blockIdx.x * blockDim.x + threadIdx.x;
  if (i < n) out[i] = f2bf(in[i]);
}

// W stored [d_in][d_out]; emit Wt bf16 [d_out][d_in] so B-frags are contiguous.
__global__ void cvt_transpose_w(const float* __restrict__ W,
                                unsigned short* __restrict__ Wt) {
  int i = blockIdx.x * blockDim.x + threadIdx.x; // over D_DIM*D_DIM
  int k = i >> 10, n = i & (D_DIM - 1);
  Wt[n * D_DIM + k] = f2bf(W[i]);
}

// ------------------------------------------------------------- QKV GEMM
// C[8192,1024] = A[8192,1024] * Bt^T ; per-wave 32x32 tile (2x2 WMMA accums).
// transposeOut=1 stores C^T (bf16 [1024][8192]) for the V matrix.
__global__ void gemm_bf16(const unsigned short* __restrict__ A,
                          const unsigned short* __restrict__ Bt,
                          unsigned short* __restrict__ C, int transposeOut) {
  const int wave = threadIdx.x >> 5;
  const int lane = threadIdx.x & 31;
  const int hi = lane >> 4;
  const int ln = lane & 15;
  const int g = blockIdx.x * 8 + wave;
  const int nt32 = D_DIM / 32; // 32 column tiles
  const int m0 = (g / nt32) * 32;
  const int n0 = (g % nt32) * 32;

  v8f acc[2][2] = {};
  for (int k0 = 0; k0 < D_DIM; k0 += 32) {
    v16bf a0 = load_a_frag(A, D_DIM, m0 + ln, k0, hi);
    v16bf a1 = load_a_frag(A, D_DIM, m0 + 16 + ln, k0, hi);
    v16bf b0 = load_b_frag(Bt + (n0 + ln) * D_DIM + k0, hi);
    v16bf b1 = load_b_frag(Bt + (n0 + 16 + ln) * D_DIM + k0, hi);
    acc[0][0] = wmma_bf16(a0, b0, acc[0][0]);
    acc[0][1] = wmma_bf16(a0, b1, acc[0][1]);
    acc[1][0] = wmma_bf16(a1, b0, acc[1][0]);
    acc[1][1] = wmma_bf16(a1, b1, acc[1][1]);
  }
  if (!transposeOut) {
    for (int i = 0; i < 2; ++i)
      for (int j = 0; j < 2; ++j)
        for (int r = 0; r < 8; ++r)
          C[(m0 + i * 16 + r + 8 * hi) * D_DIM + n0 + j * 16 + ln] =
              f2bf(acc[i][j][r]);
  } else {
    for (int i = 0; i < 2; ++i)
      for (int j = 0; j < 2; ++j) {
        ushort8 pk;
        for (int r = 0; r < 8; ++r) pk[r] = f2bf(acc[i][j][r]);
        // D rows r..r+7 are consecutive m -> one 16B store per tile per lane
        *(ushort8*)&C[(n0 + j * 16 + ln) * N_TOK + m0 + i * 16 + 8 * hi] = pk;
      }
  }
}

// ------------------------------------------------- pass A: softmax stats
// Per block: 16 query rows; waves split the 8192 keys; online (m, l).
__global__ void attn_stats(const unsigned short* __restrict__ qh,
                           const unsigned short* __restrict__ kh,
                           float* __restrict__ stats) {
  __shared__ unsigned short Qs[16 * 1032]; // padded stride: conflict-free A reads
  __shared__ float Ss[8 * 16 * 20];        // per-wave 16x16 score tile, pad 20
  __shared__ float Mred[8 * 16];
  __shared__ float Lred[8 * 16];

  const int tid = threadIdx.x;
  const int wave = tid >> 5;
  const int lane = tid & 31;
  const int hi = lane >> 4;
  const int ln = lane & 15;
  const int q0 = blockIdx.x * 16;

  for (int idx = tid * 8; idx < 16 * 1024; idx += 256 * 8) {
    int row = idx >> 10, kk = idx & 1023;
    *(ushort8*)&Qs[row * 1032 + kk] = *(const ushort8*)&qh[(q0 + row) * D_DIM + kk];
  }
  __syncthreads();

  float m_l = -INFINITY, l_l = 0.0f;
  float* myS = &Ss[wave * 16 * 20];

  for (int chunk = wave; chunk < N_TOK / 16; chunk += 8) { // 64 iters, uniform
    const int key0 = chunk * 16;
    v8f s = {};
    for (int k0 = 0; k0 < D_DIM; k0 += 32) {
      v16bf a = load_a_frag(Qs, 1032, ln, k0, hi);
      v16bf b = load_b_frag(kh + (key0 + ln) * D_DIM + k0, hi);
      s = wmma_bf16(a, b, s);
    }
    for (int r = 0; r < 8; ++r) myS[(r + 8 * hi) * 20 + ln] = s[r] * SCALE;
    __syncthreads();
    float tm = -INFINITY;
    for (int c = 0; c < 16; ++c) tm = fmaxf(tm, myS[ln * 20 + c]);
    float te = 0.0f;
    for (int c = 0; c < 16; ++c) te += __expf(myS[ln * 20 + c] - tm);
    float mn = fmaxf(m_l, tm);
    l_l = l_l * __expf(m_l - mn) + te * __expf(tm - mn);
    m_l = mn;
    __syncthreads();
  }
  Mred[wave * 16 + ln] = m_l;
  Lred[wave * 16 + ln] = l_l;
  __syncthreads();
  if (tid < 16) {
    float M = -INFINITY;
    for (int w = 0; w < 8; ++w) M = fmaxf(M, Mred[w * 16 + tid]);
    float L = 0.0f;
    for (int w = 0; w < 8; ++w) L += Lred[w * 16 + tid] * __expf(Mred[w * 16 + tid] - M);
    stats[q0 + tid] = M;
    stats[N_TOK + q0 + tid] = L;
  }
}

// ------------------------------------------------- pass B: P@V output
// Per block: 16 query rows x full 1024 D. Each wave owns a 128-wide D slice
// (8 f32 accumulators). Per 128-key chunk: each wave computes one 16x16 score
// tile (balanced), writes normalized P bf16 to LDS, then all waves run PV.
__global__ void attn_out(const unsigned short* __restrict__ qh,
                         const unsigned short* __restrict__ kh,
                         const unsigned short* __restrict__ vT,
                         const float* __restrict__ stats,
                         float* __restrict__ out) {
  __shared__ unsigned short Qs[16 * 1032];
  __shared__ unsigned short Ps[16 * 136]; // 16 x 128 P chunk, pad 136

  const int tid = threadIdx.x;
  const int wave = tid >> 5;
  const int lane = tid & 31;
  const int hi = lane >> 4;
  const int ln = lane & 15;
  const int q0 = blockIdx.x * 16;

  for (int idx = tid * 8; idx < 16 * 1024; idx += 256 * 8) {
    int row = idx >> 10, kk = idx & 1023;
    *(ushort8*)&Qs[row * 1032 + kk] = *(const ushort8*)&qh[(q0 + row) * D_DIM + kk];
  }

  float mrow[8], linv[8];
  for (int r = 0; r < 8; ++r) {
    mrow[r] = stats[q0 + r + 8 * hi];
    linv[r] = 1.0f / stats[N_TOK + q0 + r + 8 * hi];
  }
  __syncthreads();

  v8f O[8] = {};
  const int key_base = wave * 16;
  const int d0 = wave * 128;

  for (int c0 = 0; c0 < N_TOK; c0 += 128) {
    // phase 1: this wave's 16-key score tile
    const int key0 = c0 + key_base;
    v8f s = {};
    for (int k0 = 0; k0 < D_DIM; k0 += 32) {
      v16bf a = load_a_frag(Qs, 1032, ln, k0, hi);
      v16bf b = load_b_frag(kh + (key0 + ln) * D_DIM + k0, hi);
      s = wmma_bf16(a, b, s);
    }
    for (int r = 0; r < 8; ++r) {
      float p = __expf(s[r] * SCALE - mrow[r]) * linv[r];
      Ps[(r + 8 * hi) * 136 + key_base + ln] = f2bf(p);
    }
    __syncthreads();
    // phase 2: O[16 x 128-slice] += P[16x128] @ V[128 x d-slice]
    for (int kk = 0; kk < 4; ++kk) {
      v16bf a = load_a_frag(Ps, 136, ln, kk * 32, hi);
      for (int t = 0; t < 8; ++t) {
        v16bf b = load_b_frag(vT + (d0 + t * 16 + ln) * N_TOK + c0 + kk * 32, hi);
        O[t] = wmma_bf16(a, b, O[t]);
      }
    }
    __syncthreads();
  }
  for (int t = 0; t < 8; ++t)
    for (int r = 0; r < 8; ++r)
      out[(q0 + r + 8 * hi) * D_DIM + d0 + t * 16 + ln] = O[t][r];
}

// ---------------------------------------------------------------- launcher
extern "C" void kernel_launch(void* const* d_in, const int* in_sizes, int n_in,
                              void* d_out, int out_size, void* d_ws, size_t ws_size,
                              hipStream_t stream) {
  const float* x  = (const float*)d_in[0];
  const float* Wq = (const float*)d_in[1];
  const float* Wk = (const float*)d_in[2];
  const float* Wv = (const float*)d_in[3];
  float* out = (float*)d_out;
  char* ws = (char*)d_ws;

  const size_t SZ_XH = (size_t)N_TOK * D_DIM * 2; // 16 MB bf16 matrix
  const size_t SZ_WT = (size_t)D_DIM * D_DIM * 2; // 2 MB bf16 weight
  unsigned short* xh  = (unsigned short*)(ws);
  unsigned short* wqt = (unsigned short*)(ws + SZ_XH);
  unsigned short* wkt = (unsigned short*)(ws + SZ_XH + SZ_WT);
  unsigned short* wvt = (unsigned short*)(ws + SZ_XH + 2 * SZ_WT);
  unsigned short* qh  = (unsigned short*)(ws + SZ_XH + 3 * SZ_WT);
  unsigned short* kh  = (unsigned short*)(ws + 2 * SZ_XH + 3 * SZ_WT);
  unsigned short* vT  = (unsigned short*)(ws + 3 * SZ_XH + 3 * SZ_WT);
  float* stats        = (float*)(ws + 4 * SZ_XH + 3 * SZ_WT); // 2*N floats

  cvt_f32_bf16<<<(N_TOK * D_DIM) / 256, 256, 0, stream>>>(x, xh, N_TOK * D_DIM);
  cvt_transpose_w<<<(D_DIM * D_DIM) / 256, 256, 0, stream>>>(Wq, wqt);
  cvt_transpose_w<<<(D_DIM * D_DIM) / 256, 256, 0, stream>>>(Wk, wkt);
  cvt_transpose_w<<<(D_DIM * D_DIM) / 256, 256, 0, stream>>>(Wv, wvt);

  const int gemmBlocks = (N_TOK / 32) * (D_DIM / 32) / 8; // 1024 blocks of 8 waves
  gemm_bf16<<<gemmBlocks, 256, 0, stream>>>(xh, wqt, qh, 0);
  gemm_bf16<<<gemmBlocks, 256, 0, stream>>>(xh, wkt, kh, 0);
  gemm_bf16<<<gemmBlocks, 256, 0, stream>>>(xh, wvt, vT, 1);

  attn_stats<<<N_TOK / 16, 256, 0, stream>>>(qh, kh, stats);
  attn_out<<<N_TOK / 16, 256, 0, stream>>>(qh, kh, vT, stats, out);
}
